// HierESN_90486370992872
// MI455X (gfx1250) — matrix operand order repaired and checked by reading, not compiled, not run on hardware
//
#include <hip/hip_runtime.h>
#include <math.h>

// ---------------------------------------------------------------------------
// Hierarchical leaky ESN on MI455X (gfx1250, wave32, WMMA).
//   T=1000, N_IN=4096, R1=R2=4000, alpha=0.5
//   1) Weights fp32 -> bf16, packed in WMMA B-fragment tile order (one time).
//      bf16 weights (~129MB) fit in the 192MB L2 -> sequential loop is
//      L2-resident instead of HBM-bound.
//   2) U1 = X @ Win1 hoisted out of the scan as a bf16 WMMA GEMM
//      (X zero-padded to 1008 rows so the k-loop has no select VALU).
//   3) Persistent kernel, 2 grid syncs/step, load-balanced:
//        Phase A: v2.W2 split-K over all 500 waves (+ elementwise v1n).
//        Phase B: v1n.Win2 (waves 0-249)  ||  v1n.W1 -> z1buf for t+1 (250-499).
//      GEMVs are WMMA with the bf16 state vector broadcast across A rows;
//      A-fragments are pure aligned b128 loads (bf16 shadow state).
// ---------------------------------------------------------------------------

typedef __attribute__((ext_vector_type(16))) __bf16 v16bf;
typedef __attribute__((ext_vector_type(8)))  __bf16 v8bf;
typedef __attribute__((ext_vector_type(8)))  float  v8f;

#define T_STEPS 1000
#define N_IN    4096
#define RDIM    4000
#define KT_R    125      // 4000 / 32
#define NT_R    250      // 4000 / 16
#define KT_IN   128      // 4096 / 32
#define MT_T    63       // ceil(1000 / 16)
#define T_PAD   (MT_T * 16)   // 1008 zero-padded X rows
#define ALPHA   0.5f

__device__ __forceinline__ __bf16 to_bf16(float f) {
  unsigned int u = __builtin_bit_cast(unsigned int, f);
  unsigned int r = u + 0x7FFFu + ((u >> 16) & 1u);     // round-to-nearest-even
  unsigned short h = (unsigned short)(r >> 16);
  return __builtin_bit_cast(__bf16, h);
}

#define WMMA_BF16(a, b, c) \
  __builtin_amdgcn_wmma_f32_16x16x32_bf16(false, (a), false, (b), (short)0, (c), false, false)

// ---------------------------------------------------------------------------
// Pack W[K,N] (row-major fp32) into WMMA B-fragment tiles:
// tile (kt,nt) is 32x16 (KxN); per lane: 16 contiguous bf16 covering
// K = kt*32 + (lane<16 ? 0 : 8) + j + (j<8 ? 0 : 8), N = nt*16 + (lane&15).
// => B-fragment load in the hot loop is 32B contiguous per lane (2x b128).
// ---------------------------------------------------------------------------
__global__ void __launch_bounds__(256)
pack_bf16_kernel(const float* __restrict__ W, __bf16* __restrict__ P,
                 int KT, int NT, int N) {
  const int idx   = blockIdx.x * blockDim.x + threadIdx.x;
  const int total = KT * NT * 32;
  if (idx >= total) return;
  const int lane = idx & 31;
  const int tile = idx >> 5;
  const int nt   = tile % NT;
  const int kt   = tile / NT;
  const int n    = nt * 16 + (lane & 15);
  const int kb   = kt * 32 + ((lane >= 16) ? 8 : 0);
  __bf16* dst = P + (size_t)idx * 16;
#pragma unroll
  for (int j = 0; j < 16; ++j)
    dst[j] = to_bf16(W[(size_t)(kb + j + ((j >= 8) ? 8 : 0)) * N + n]);
}

// fp32 -> bf16 with zero tail padding (i >= nsrc -> 0)
__global__ void __launch_bounds__(256)
convert_pad_bf16_kernel(const float* __restrict__ src, __bf16* __restrict__ dst,
                        int nsrc, int ntotal) {
  const int i = blockIdx.x * blockDim.x + threadIdx.x;
  if (i < ntotal) dst[i] = (i < nsrc) ? to_bf16(src[i]) : to_bf16(0.0f);
}

// A-fragment (16x32 bf16, MxK) from a *linear bf16 vector*:
// lanes 0-15 carry K {0..7, 16..23}, lanes 16-31 carry K {8..15, 24..31}.
// Two aligned 16B loads + register concat; no conversion VALU.
__device__ __forceinline__ v16bf load_afrag(const __bf16* __restrict__ vbf,
                                            int kb, int koff) {
  const v8bf lo = *reinterpret_cast<const v8bf*>(vbf + kb + koff);
  const v8bf hi = *reinterpret_cast<const v8bf*>(vbf + kb + 16 + koff);
  return __builtin_shufflevector(lo, hi, 0, 1, 2, 3, 4, 5, 6, 7,
                                 8, 9, 10, 11, 12, 13, 14, 15);
}

// 16-column GEMV tile over kt in [kt0, kt1): dual accumulators stretch the
// WMMA->WMMA dependency distance past the bf16 hazard window.
__device__ __forceinline__ v8f gemv_tile(const __bf16* __restrict__ vbf,
                                         const __bf16* __restrict__ Bpack,
                                         int ntile, int lane, int kt0, int kt1) {
  const int koff = (lane < 16) ? 0 : 8;
  const __bf16* bp = Bpack + (((size_t)ntile * 32) + lane) * 16;
  const size_t kstride = (size_t)NT_R * 512;   // elements per kt of packed W
  v8f c0 = {}, c1 = {};
  int kt = kt0;
  for (; kt + 2 <= kt1; kt += 2) {
    const v16bf a0 = load_afrag(vbf, kt * 32, koff);
    const v16bf b0 = *reinterpret_cast<const v16bf*>(bp + (size_t)kt * kstride);
    const v16bf a1 = load_afrag(vbf, (kt + 1) * 32, koff);
    const v16bf b1 = *reinterpret_cast<const v16bf*>(bp + (size_t)(kt + 1) * kstride);
    c0 = WMMA_BF16(a0, b0, c0);
    c1 = WMMA_BF16(a1, b1, c1);
  }
  if (kt < kt1) {
    const v16bf a0 = load_afrag(vbf, kt * 32, koff);
    const v16bf b0 = *reinterpret_cast<const v16bf*>(bp + (size_t)kt * kstride);
    c0 = WMMA_BF16(a0, b0, c0);
  }
  return c0 + c1;
}

// ---------------------------------------------------------------------------
// U1 = X(1008x4096, zero-padded) @ Win1(4096x4000) -- one 16x16 tile per wave.
// No per-iteration select: padded rows are zeros in memory.
// ---------------------------------------------------------------------------
__global__ void __launch_bounds__(256)
u1_gemm_kernel(const __bf16* __restrict__ Xbf, const __bf16* __restrict__ Win1p,
               float* __restrict__ U1) {
  const int lane = threadIdx.x & 31;
  const int wg   = blockIdx.x * (blockDim.x >> 5) + (threadIdx.x >> 5);
  if (wg >= MT_T * NT_R) return;
  const int mt = wg / NT_R, nt = wg % NT_R;
  const int m0 = mt * 16,   n0 = nt * 16;
  const int koff = (lane < 16) ? 0 : 8;
  const __bf16* xr = Xbf + (size_t)(m0 + (lane & 15)) * N_IN;
  const __bf16* bp = Win1p + (((size_t)nt * 32) + lane) * 16;
  const size_t kstride = (size_t)NT_R * 512;
  v8f c0 = {}, c1 = {};
  for (int kt = 0; kt < KT_IN; kt += 2) {
    const v16bf a0 = load_afrag(xr, kt * 32, koff);
    const v16bf b0 = *reinterpret_cast<const v16bf*>(bp + (size_t)kt * kstride);
    const v16bf a1 = load_afrag(xr, (kt + 1) * 32, koff);
    const v16bf b1 = *reinterpret_cast<const v16bf*>(bp + (size_t)(kt + 1) * kstride);
    c0 = WMMA_BF16(a0, b0, c0);
    c1 = WMMA_BF16(a1, b1, c1);
  }
  const v8f c = c0 + c1;
  // D layout: VGPR r -> M = r (+8 for lanes 16-31), N = lane&15
#pragma unroll
  for (int r = 0; r < 8; ++r) {
    const int row = m0 + r + ((lane >= 16) ? 8 : 0);
    const int col = n0 + (lane & 15);
    if (row < T_STEPS) U1[(size_t)row * RDIM + col] = c[r];
  }
}

// ---------------------------------------------------------------------------
// Device-wide generation barrier (persistent-grid sync)
// ---------------------------------------------------------------------------
__device__ __forceinline__ void grid_sync(unsigned int* cnt,
                                          volatile unsigned int* gen,
                                          unsigned int nblocks) {
  __syncthreads();
  if (threadIdx.x == 0) {
    const unsigned int g = *gen;
    __threadfence();
    if (atomicAdd(cnt, 1u) == nblocks - 1u) {
      *cnt = 0u;
      __threadfence();
      atomicAdd((unsigned int*)gen, 1u);
    } else {
      while (*gen == g) { __builtin_amdgcn_s_sleep(1); }
    }
    __threadfence();
  }
  __syncthreads();
}

__global__ void init_sync_kernel(unsigned int* sync) {
  if (threadIdx.x < 2) sync[threadIdx.x] = 0u;
}

// ---------------------------------------------------------------------------
// Persistent sequential loop. 63 blocks x 256 threads = 504 wave32s.
//
// Phase A(t):  waves 0..499 : z2part[h][tile] = v2(t-1) . W2   (split-K 2-way)
//              waves 0..249 : v1n(t) = leak(v1, tanh(z1buf + U1[t]))
//                             -> out, v1bf            (elementwise, cheap)
// sync
// Phase B(t):  waves 0..249 : v2n(t) = leak(v2, tanh(v1n.Win2 + sum z2part))
//                             -> out, v2bf
//              waves 250..499: z1buf = v1n(t) . W1    (pipelined for t+1)
// sync
//
// Critical path/step ~ 63 + 125 k-tiles (vs 250 unbalanced). Every buffer has
// its producer and consumer separated by one sync -> no double buffering.
// ---------------------------------------------------------------------------
__global__ void __launch_bounds__(256)
esn_loop_kernel(const float* __restrict__ U1,
                const __bf16* __restrict__ W1p,
                const __bf16* __restrict__ Win2p,
                const __bf16* __restrict__ W2p,
                float* __restrict__ out,
                float* __restrict__ z1buf,    // [RDIM]
                float* __restrict__ z2part,   // [2][RDIM]
                __bf16* __restrict__ v1bf,    // [RDIM]
                __bf16* __restrict__ v2bf,    // [RDIM]
                unsigned int* __restrict__ sync) {
  const int lane = threadIdx.x & 31;
  const int wg   = blockIdx.x * (blockDim.x >> 5) + (threadIdx.x >> 5);
  unsigned int* cnt = sync;
  volatile unsigned int* gen = sync + 1;
  const unsigned int nb = gridDim.x;

  for (int t = 0; t < T_STEPS; ++t) {
    // ---------------- Phase A ----------------
    if (wg < 2 * NT_R) {                   // v2 . W2, split-K halves
      const int nt = wg >> 1, h = wg & 1;
      v8f c = {};
      if (t > 0) c = gemv_tile(v2bf, W2p, nt, lane, h ? 63 : 0, h ? KT_R : 63);
      if (lane < 16) z2part[(size_t)h * RDIM + nt * 16 + lane] = c[0];
    }
    if (wg < NT_R && lane < 16) {          // v1n finalize (elementwise)
      const int col = wg * 16 + lane;
      const float z   = ((t > 0) ? z1buf[col] : 0.0f) + U1[(size_t)t * RDIM + col];
      const float vp  = (t > 0) ? out[(size_t)(t - 1) * (2 * RDIM) + col] : 0.0f;
      const float v1n = (1.0f - ALPHA) * vp + ALPHA * tanhf(z);
      out[(size_t)t * (2 * RDIM) + col] = v1n;
      v1bf[col] = to_bf16(v1n);
    }
    grid_sync(cnt, gen, nb);

    // ---------------- Phase B ----------------
    if (wg < NT_R) {                       // v2 update, tile wg
      const v8f c = gemv_tile(v1bf, Win2p, wg, lane, 0, KT_R);
      if (lane < 16) {
        const int col = wg * 16 + lane;
        const float z   = c[0] + z2part[col] + z2part[(size_t)RDIM + col];
        const float vp  = (t > 0) ? out[(size_t)(t - 1) * (2 * RDIM) + RDIM + col] : 0.0f;
        const float v2n = (1.0f - ALPHA) * vp + ALPHA * tanhf(z);
        out[(size_t)t * (2 * RDIM) + RDIM + col] = v2n;
        v2bf[col] = to_bf16(v2n);
      }
    } else if (wg < 2 * NT_R) {            // v1n . W1 -> z1buf (for step t+1)
      const int nt = wg - NT_R;
      const v8f c = gemv_tile(v1bf, W1p, nt, lane, 0, KT_R);
      if (lane < 16) z1buf[nt * 16 + lane] = c[0];
    }
    grid_sync(cnt, gen, nb);
  }
}

// ---------------------------------------------------------------------------
extern "C" void kernel_launch(void* const* d_in, const int* in_sizes, int n_in,
                              void* d_out, int out_size, void* d_ws, size_t ws_size,
                              hipStream_t stream) {
  (void)in_sizes; (void)n_in; (void)out_size; (void)ws_size;
  const float* x    = (const float*)d_in[0];   // [T, N_IN]
  const float* Win1 = (const float*)d_in[1];   // [N_IN, R1]
  const float* W1   = (const float*)d_in[2];   // [R1, R1]
  const float* Win2 = (const float*)d_in[3];   // [R1, R2]
  const float* W2   = (const float*)d_in[4];   // [R2, R2]
  float* out = (float*)d_out;                  // [T, R1+R2]

  // Workspace carve-out (256B aligned)
  char* ws = (char*)d_ws;
  size_t off = 0;
  auto carve = [&](size_t bytes) -> void* {
    void* p = (void*)(ws + off);
    off += (bytes + 255) & ~(size_t)255;
    return p;
  };
  __bf16* Win1p = (__bf16*)carve((size_t)KT_IN * NT_R * 32 * 16 * 2); // 32.8 MB
  __bf16* W1p   = (__bf16*)carve((size_t)KT_R  * NT_R * 32 * 16 * 2); // 32.0 MB
  __bf16* Win2p = (__bf16*)carve((size_t)KT_R  * NT_R * 32 * 16 * 2); // 32.0 MB
  __bf16* W2p   = (__bf16*)carve((size_t)KT_R  * NT_R * 32 * 16 * 2); // 32.0 MB
  float*  U1    = (float*) carve((size_t)T_STEPS * RDIM * 4);         // 16.0 MB
  __bf16* Xbf   = (__bf16*)carve((size_t)T_PAD * N_IN * 2);           //  8.3 MB
  float*  z1buf = (float*) carve((size_t)RDIM * 4);
  float*  z2part= (float*) carve((size_t)2 * RDIM * 4);
  __bf16* v1bf  = (__bf16*)carve((size_t)RDIM * 2);
  __bf16* v2bf  = (__bf16*)carve((size_t)RDIM * 2);
  unsigned int* sync = (unsigned int*)carve(256);

  // 1) One-time fp32 -> bf16 packing / padded conversion.
  {
    int tot = KT_IN * NT_R * 32;
    pack_bf16_kernel<<<(tot + 255) / 256, 256, 0, stream>>>(Win1, Win1p, KT_IN, NT_R, RDIM);
    tot = KT_R * NT_R * 32;
    pack_bf16_kernel<<<(tot + 255) / 256, 256, 0, stream>>>(W1,   W1p,   KT_R, NT_R, RDIM);
    pack_bf16_kernel<<<(tot + 255) / 256, 256, 0, stream>>>(Win2, Win2p, KT_R, NT_R, RDIM);
    pack_bf16_kernel<<<(tot + 255) / 256, 256, 0, stream>>>(W2,   W2p,   KT_R, NT_R, RDIM);
    const int nx = T_STEPS * N_IN, nxp = T_PAD * N_IN;
    convert_pad_bf16_kernel<<<(nxp + 255) / 256, 256, 0, stream>>>(x, Xbf, nx, nxp);
  }

  // 2) Time-parallel input projection U1 = X @ Win1 (WMMA GEMM).
  {
    const int tiles = MT_T * NT_R;               // 15750 tiles, 8 waves/block
    u1_gemm_kernel<<<(tiles + 7) / 8, 256, 0, stream>>>(Xbf, Win1p, U1);
  }

  // 3) Sequential recurrence, persistent grid + device-wide barrier.
  init_sync_kernel<<<1, 32, 0, stream>>>(sync);
  esn_loop_kernel<<<63, 256, 0, stream>>>(U1, W1p, Win2p, W2p, out,
                                          z1buf, z2part, v1bf, v2bf, sync);
}
